// MinimalRNNCell_43336220016740
// MI455X (gfx1250) — matrix operand order repaired and checked by reading, not compiled, not run on hardware
//
#include <hip/hip_runtime.h>

typedef __attribute__((ext_vector_type(16))) __bf16 v16bf;
typedef __attribute__((ext_vector_type(8)))  __bf16 v8bf;
typedef __attribute__((ext_vector_type(2)))  __bf16 v2bf;
typedef __attribute__((ext_vector_type(8)))  float  v8f;
typedef __attribute__((ext_vector_type(4)))  float  v4f;

#if defined(__gfx1250__) && __has_builtin(__builtin_amdgcn_tensor_load_to_lds)
#define HAVE_TDM 1
typedef unsigned int u32x4 __attribute__((ext_vector_type(4)));
typedef int          i32x8 __attribute__((ext_vector_type(8)));
typedef int          i32x4 __attribute__((ext_vector_type(4)));
#else
#define HAVE_TDM 0
#endif

#define B_  32
#define T_  1024
#define D_  512
#define U_  1024
#define NWG_SCAN 16          // one 64-col slab of U per WGP (128KB bf16 in LDS)

// ---------------------------------------------------------------------------
// Fragment-order layouts (wave32 WMMA, 16x16x32 bf16):
//  A (16x32, MxK): lane = m + 16*(k/16),    elem idx = k%16
//  B (32x16, KxN): lane = n + 16*(k/16),    elem idx = k%16
//  C/D (16x16 f32): VGPR i -> M = i + 8*(lane/16), N = lane%16
// h_bf in GLOBAL memory is kept in A-fragment order so the per-step stage is
// a flat 64KB copy: lower 32KB via one TENSOR_LOAD_TO_LDS descriptor (TDM,
// TENSORcnt), upper 32KB via GLOBAL_LOAD_ASYNC_TO_LDS_B128 (ASYNCcnt).
// ---------------------------------------------------------------------------

__device__ inline v8bf cvt8(v4f a, v4f b) {
  v8bf r;
  r[0]=(__bf16)a[0]; r[1]=(__bf16)a[1]; r[2]=(__bf16)a[2]; r[3]=(__bf16)a[3];
  r[4]=(__bf16)b[0]; r[5]=(__bf16)b[1]; r[6]=(__bf16)b[2]; r[7]=(__bf16)b[3];
  return r;
}

// A-fragment element offset (in bf16 elements) for h[b, col] in a 32x1024 matrix
__device__ inline size_t hfrag_off(int b, int col) {
  int kl = col & 31;
  return (size_t)(((b >> 4) * 32) + (col >> 5)) * 512
       + ((b & 15) + ((kl >> 4) << 4)) * 16
       + (kl & 15);
}

// ---------------- init: h_bf(frag order) = bf16(h0), reset barrier ---------
__global__ void rnn_init(const float* __restrict__ h0, __bf16* __restrict__ h_bf,
                         unsigned* __restrict__ ctr) {
  int i = blockIdx.x * blockDim.x + threadIdx.x;
  if (i < B_ * U_) {
    int b = i >> 10, col = i & (U_ - 1);
    h_bf[hfrag_off(b, col)] = (__bf16)h0[i];
  }
  if (i == 0) *ctr = 0u;
}

// ---------------- phase 1: OUT[b,t,:] = x[b,t,:] @ W  (bf16 WMMA) ----------
__global__ __launch_bounds__(256) void xw_gemm(const float* __restrict__ X,
                                               const float* __restrict__ W,
                                               float* __restrict__ OUT) {
  __shared__ __align__(32) char smem[16384];   // A frags 8KB | B frags 8KB
  const int tid  = threadIdx.x;
  const int lane = tid & 31;
  const int wave = tid >> 5;          // 8 waves
  const int wr   = wave >> 2;         // 0..1  (64 rows each)
  const int wc   = wave & 3;          // 0..3  (32 cols each)
  const int rowBase = blockIdx.x * 128;
  const int colBase = blockIdx.y * 128;

  // staging roles
  const int am  = tid >> 1;           // A: row 0..127
  const int akh = (tid & 1) << 4;     // A: k-half 0 / 16
  const int bk  = (tid >> 4) << 1;    // B: even k 0..30 (thread does k,k+1)
  const int bn0 = (tid & 15) << 3;    // B: 8-col group

  v8f acc[4][2] = {};

  for (int kt = 0; kt < D_ / 32; ++kt) {
    const int k0 = kt * 32;
    __syncthreads();
    // ---- stage A tile (128x32 fp32 -> bf16 fragments, 2x b128 stores) ----
    {
      const float* src = X + (size_t)(rowBase + am) * D_ + k0 + akh;
      v4f f0 = *(const v4f*)(src + 0);
      v4f f1 = *(const v4f*)(src + 4);
      v4f f2 = *(const v4f*)(src + 8);
      v4f f3 = *(const v4f*)(src + 12);
      v8bf lo = cvt8(f0, f1);
      v8bf hi = cvt8(f2, f3);
      char* ab = smem + (am >> 4) * 1024;
      *(v8bf*)(ab + ((am & 15) + akh) * 32     ) = lo;
      *(v8bf*)(ab + ((am & 15) + akh) * 32 + 16) = hi;
    }
    // ---- stage B tile (32x128 fp32 -> bf16, packed k-pair b32 stores) ----
    {
      const float* s0 = W + (size_t)(k0 + bk) * U_ + colBase + bn0;
      const float* s1 = s0 + U_;
      v4f r00 = *(const v4f*)(s0), r01 = *(const v4f*)(s0 + 4);
      v4f r10 = *(const v4f*)(s1), r11 = *(const v4f*)(s1 + 4);
      char* bb = smem + 8192 + (bn0 >> 4) * 1024
               + ((bn0 & 15) + ((bk >> 4) << 4)) * 32 + (bk & 15) * 2;
      #pragma unroll
      for (int i = 0; i < 4; ++i) {
        v2bf p0; p0[0] = (__bf16)r00[i]; p0[1] = (__bf16)r10[i];
        *(v2bf*)(bb + i * 32) = p0;
        v2bf p1; p1[0] = (__bf16)r01[i]; p1[1] = (__bf16)r11[i];
        *(v2bf*)(bb + (i + 4) * 32) = p1;
      }
    }
    __syncthreads();
    if (kt + 1 < D_ / 32) {            // global_prefetch_b8 next tiles
      __builtin_prefetch(X + (size_t)(rowBase + am) * D_ + k0 + 32 + akh, 0, 1);
      __builtin_prefetch(W + (size_t)(k0 + 32 + bk) * U_ + colBase + bn0, 0, 1);
    }
    // ---- compute: 8 wmma per wave per k-step ----
    v16bf Bf[2];
    #pragma unroll
    for (int n = 0; n < 2; ++n)
      Bf[n] = *(const v16bf*)(smem + 8192 + (wc * 2 + n) * 1024 + lane * 32);
    #pragma unroll
    for (int a = 0; a < 4; ++a) {
      v16bf Af = *(const v16bf*)(smem + (wr * 4 + a) * 1024 + lane * 32);
      #pragma unroll
      for (int n = 0; n < 2; ++n)
        acc[a][n] = __builtin_amdgcn_wmma_f32_16x16x32_bf16(
            false, Af, false, Bf[n], (short)0, acc[a][n], false, false);
    }
  }
  // ---- write 64x32 per wave ----
  const int mh = (lane >> 4) * 8;
  const int nc = lane & 15;
  #pragma unroll
  for (int a = 0; a < 4; ++a)
    #pragma unroll
    for (int n = 0; n < 2; ++n) {
      const int col  = colBase + (wc * 2 + n) * 16 + nc;
      const int row0 = rowBase + (wr * 4 + a) * 16 + mh;
      #pragma unroll
      for (int i = 0; i < 8; ++i)
        OUT[(size_t)(row0 + i) * U_ + col] = acc[a][n][i];
    }
}

// ---------------- phase 2: persistent scan, U slab pinned in LDS -----------
__global__ __launch_bounds__(256) void rnn_scan(const float* __restrict__ Urec,
                                                float* __restrict__ OUT,
                                                __bf16* __restrict__ h_bf,
                                                unsigned* __restrict__ ctr) {
  extern __shared__ __align__(32) char smem[]; // [0,128K): U frags  [128K,192K): h frags
  const int tid  = threadIdx.x;
  const int lane = tid & 31;
  const int wave = tid >> 5;
  const int colBase = blockIdx.x * 64;   // this WG's 64 columns of U
  const int mtile = wave >> 2;           // 0..1 (batch rows 0-15 / 16-31)
  const int ntile = wave & 3;            // 0..3 (16-col groups)

  // ---- one-time: U slab (1024x64) -> bf16 B-fragments (packed b32 stores) ----
  for (int it = 0; it < 16; ++it) {
    int p  = tid + it * 256;             // 4096 (k-pair, 8-col) tasks
    int k  = (p >> 3) << 1;              // even k 0..1022
    int n0 = (p & 7) << 3;               // 0..56
    const float* s0 = Urec + (size_t)k * U_ + colBase + n0;
    const float* s1 = s0 + U_;
    v4f r00 = *(const v4f*)(s0), r01 = *(const v4f*)(s0 + 4);
    v4f r10 = *(const v4f*)(s1), r11 = *(const v4f*)(s1 + 4);
    char* bb = smem + (size_t)((k >> 5) * 4 + (n0 >> 4)) * 1024
             + ((n0 & 15) + (((k & 31) >> 4) << 4)) * 32 + (k & 15) * 2;
    #pragma unroll
    for (int i = 0; i < 4; ++i) {
      v2bf p0; p0[0] = (__bf16)r00[i]; p0[1] = (__bf16)r10[i];
      *(v2bf*)(bb + i * 32) = p0;
      v2bf p1; p1[0] = (__bf16)r01[i]; p1[1] = (__bf16)r11[i];
      *(v2bf*)(bb + (i + 4) * 32) = p1;
    }
  }

  const unsigned           ldsHBase = (unsigned)(uintptr_t)(smem + 131072);
  const unsigned long long gHBase   = (unsigned long long)(uintptr_t)h_bf;

#if HAVE_TDM
  // TDM descriptor: copy lower 32KB (4096 x 8B elements) h_bf -> LDS h-frags.
  // Loop-invariant: built once, reissued every step.
  u32x4 g0;
  g0[0] = 1u;                                        // count=1, load, no gather
  g0[1] = ldsHBase;                                  // LDS dest (bytes)
  g0[2] = (unsigned)(gHBase & 0xffffffffull);        // global addr [31:0]
  g0[3] = (unsigned)(gHBase >> 32) | (2u << 30);     // addr[56:32] | type=2
  i32x8 g1;
  g1[0] = 0x00030000;                                // data_size=8B, mask=0
  g1[1] = (int)(4096u << 16);                        // tensor_dim0[15:0]
  g1[2] = (int)(1u << 16);                           // dim0 hi | tensor_dim1=1
  g1[3] = (int)(4096u << 16);                        // tile_dim0=4096
  g1[4] = 1;                                         // tile_dim1=1, tile_dim2=0
  g1[5] = 4096;                                      // tensor_dim0_stride
  g1[6] = 0; g1[7] = 0;
  i32x4 gz4 = {0, 0, 0, 0};
  i32x8 gz8 = {0, 0, 0, 0, 0, 0, 0, 0};
#endif

  for (int t = 0; t < T_; ++t) {
    __syncthreads();
#if HAVE_TDM
    // ---- lower 32KB via Tensor Data Mover (one descriptor, wave 0) ----
    if (wave == 0)
      __builtin_amdgcn_tensor_load_to_lds(g0, g1, gz4, gz4, gz8, 0);
    // ---- upper 32KB via async-load pipe (all waves) ----
    #pragma unroll
    for (int it = 0; it < 8; ++it) {
      unsigned off = 32768u + (unsigned)(tid * 16 + it * 4096);
      unsigned laddr = ldsHBase + off;
      unsigned long long gaddr = gHBase + off;
      asm volatile("global_load_async_to_lds_b128 %0, %1, off"
                   :: "v"(laddr), "v"(gaddr) : "memory");
    }
    asm volatile("s_wait_asynccnt 0x0" ::: "memory");
    if (wave == 0) __builtin_amdgcn_s_wait_tensorcnt(0);
#else
    // ---- full 64KB via async-load pipe ----
    #pragma unroll
    for (int it = 0; it < 16; ++it) {
      unsigned off = (unsigned)(tid * 16 + it * 4096);
      unsigned laddr = ldsHBase + off;
      unsigned long long gaddr = gHBase + off;
      asm volatile("global_load_async_to_lds_b128 %0, %1, off"
                   :: "v"(laddr), "v"(gaddr) : "memory");
    }
    asm volatile("s_wait_asynccnt 0x0" ::: "memory");
#endif
    __syncthreads();
    // ---- acc = xw[:,t,cols]; acc += h @ U_slab (32 wmma / wave) ----
    const int mh  = (lane >> 4) * 8;
    const int nc  = lane & 15;
    const int col = colBase + ntile * 16 + nc;
    v8f acc;
    #pragma unroll
    for (int i = 0; i < 8; ++i) {
      int b = mtile * 16 + mh + i;
      acc[i] = OUT[((size_t)b * T_ + t) * U_ + col];
    }
    if (t + 1 < T_) {                    // prefetch next timestep's xw gather
      #pragma unroll
      for (int i = 0; i < 8; ++i) {
        int b = mtile * 16 + mh + i;
        __builtin_prefetch(OUT + ((size_t)b * T_ + t + 1) * U_ + col, 0, 1);
      }
    }
    #pragma unroll 4
    for (int ks = 0; ks < 32; ++ks) {
      v16bf Af = *(const v16bf*)(smem + 131072 + (size_t)(mtile * 32 + ks) * 1024 + lane * 32);
      v16bf Bf = *(const v16bf*)(smem + (size_t)(ks * 4 + ntile) * 1024 + lane * 32);
      acc = __builtin_amdgcn_wmma_f32_16x16x32_bf16(
          false, Af, false, Bf, (short)0, acc, false, false);
    }
    // ---- h_t out: fp32 to OUT, bf16 mirror (frag order) for next step ----
    #pragma unroll
    for (int i = 0; i < 8; ++i) {
      int b = mtile * 16 + mh + i;
      float v = acc[i];
      OUT[((size_t)b * T_ + t) * U_ + col] = v;
      h_bf[hfrag_off(b, col)] = (__bf16)v;
    }
    // ---- device-scope step barrier (release h_bf, acquire others') ----
    __syncthreads();
    if (tid == 0) {
      __hip_atomic_fetch_add(ctr, 1u, __ATOMIC_RELEASE, __HIP_MEMORY_SCOPE_AGENT);
      unsigned target = (unsigned)(NWG_SCAN * (t + 1));
      while (__hip_atomic_load(ctr, __ATOMIC_ACQUIRE, __HIP_MEMORY_SCOPE_AGENT) < target)
        __builtin_amdgcn_s_sleep(2);
    }
    __syncthreads();
  }
}

// ---------------------------------------------------------------------------
extern "C" void kernel_launch(void* const* d_in, const int* in_sizes, int n_in,
                              void* d_out, int out_size, void* d_ws, size_t ws_size,
                              hipStream_t stream) {
  const float* x  = (const float*)d_in[0];
  const float* h0 = (const float*)d_in[1];
  const float* W  = (const float*)d_in[2];
  const float* Ur = (const float*)d_in[3];
  float* out = (float*)d_out;

  __bf16*   h_bf = (__bf16*)d_ws;                                   // 64 KB, frag order
  unsigned* ctr  = (unsigned*)((char*)d_ws + B_ * U_ * sizeof(__bf16));

  (void)hipFuncSetAttribute(reinterpret_cast<const void*>(rnn_scan),
                            hipFuncAttributeMaxDynamicSharedMemorySize, 196608);

  rnn_init<<<dim3((B_ * U_ + 255) / 256), dim3(256), 0, stream>>>(h0, h_bf, ctr);
  xw_gemm<<<dim3(B_ * T_ / 128, U_ / 128), dim3(256), 0, stream>>>(x, W, out);
  rnn_scan<<<dim3(NWG_SCAN), dim3(256), 196608, stream>>>(Ur, out, h_bf, ctr);
}
